// MultiHeadAttention_35356170781390
// MI455X (gfx1250) — matrix operand order repaired and checked by reading, not compiled, not run on hardware
//
#include <hip/hip_runtime.h>
#include <hip/hip_bf16.h>
#include <cstdint>

// ---------------------------------------------------------------------------
// MHA forward for MI455X (gfx1250, wave32, WMMA + Tensor Data Mover).
//  B=4, S=2048, D_MODEL=512, H=8, Dk=Dv=64.
//  d_out = [out: B*S*512 f32][attn: B*H*S*S f32]
// ---------------------------------------------------------------------------

#define B_    4
#define S_    2048
#define DM_   512
#define H_    8
#define DKV_  64
#define SCALE 0.125f      // 1/sqrt(64)
#define LN_EPS 1e-5f

typedef __attribute__((ext_vector_type(16))) _Float16 v16h;
typedef __attribute__((ext_vector_type(8)))  float    v8f;
typedef __attribute__((ext_vector_type(4)))  unsigned int u32x4;
typedef __attribute__((ext_vector_type(8)))  int          i32x8;
typedef __attribute__((ext_vector_type(4)))  int          i32x4;

#ifndef __has_builtin
#define __has_builtin(x) 0
#endif
#if defined(__HIP_DEVICE_COMPILE__) && __has_builtin(__builtin_amdgcn_tensor_load_to_lds)
#define HAVE_TDM 1
#else
#define HAVE_TDM 0
#endif

union Frag16 {            // 8 VGPRs = 16 halves (A or B fragment, wave32)
    v16h     v;
    uint32_t u[8];
};

__device__ __forceinline__ v8f wmma_f16(const Frag16& a, const Frag16& b, v8f c) {
    return __builtin_amdgcn_wmma_f32_16x16x32_f16(
        /*neg_a=*/false, a.v, /*neg_b=*/false, b.v,
        /*c_mod=*/(short)0, c, /*reuse_a=*/false, /*reuse_b=*/false);
}

// A-fragment K index for vgpr v, half-select hi (contiguous k = 16*hi + 2*v, matches B)
__device__ __forceinline__ int a_kidx(int v, int hi) {
    return 2 * v + 16 * hi;
}

// ---------------------------------------------------------------------------
// TDM: DMA a 2D f16 tile (tile_d1 rows x tile_d0 elems) into LDS with
// hardware padding: +1 dword after every 32 dwords -> LDS row stride 132 B.
// Descriptor layout per CDNA5 ISA.
// clang-23 toolchain: 6-arg builtin (g0, g1, g2, g3, g4, cpol).
// ---------------------------------------------------------------------------
#define TROW   66      // padded LDS row stride in halves (132 B)
#define TILE_B 8448    // 64 rows * 132 B per staged tile

#if HAVE_TDM
__device__ __forceinline__ void tdm_load_2d(
    unsigned lds_off, const void* gptr,
    unsigned tile_d0, unsigned tile_d1,
    unsigned tensor_d0, unsigned tensor_d1, unsigned stride0)
{
    unsigned long long ga = (unsigned long long)(uintptr_t)gptr;
    u32x4 g0;
    g0[0] = 1u;                                            // count=1, user D#
    g0[1] = lds_off;                                       // lds_addr (bytes)
    g0[2] = (unsigned)ga;                                  // global_addr[31:0]
    g0[3] = (unsigned)((ga >> 32) & 0x01FFFFFFull)         // global_addr[56:32]
          | (2u << 30);                                    // type=2 (image)
    i32x8 g1;
    g1[0] = (int)((1u << 16)        // data_size = 1 -> 2 bytes
          |       (1u << 20)        // pad_enable
          |       (4u << 22));      // pad_interval = 32 dwords; pad_amount = 1 dword
    g1[1] = (int)((tensor_d0 & 0xFFFFu) << 16);            // tensor_dim0[15:0]
    g1[2] = (int)((tensor_d0 >> 16) | ((tensor_d1 & 0xFFFFu) << 16));
    g1[3] = (int)((tensor_d1 >> 16) | ((tile_d0 & 0xFFFFu) << 16));
    g1[4] = (int)(tile_d1 & 0xFFFFu);                      // tile_dim1 (tile_dim2=0)
    g1[5] = (int)stride0;                                  // tensor_dim0_stride lo
    g1[6] = 0;
    g1[7] = 0;
    i32x4 z4 = {0, 0, 0, 0};
    i32x8 z8 = {0, 0, 0, 0, 0, 0, 0, 0};
    __builtin_amdgcn_tensor_load_to_lds(g0, g1, z4, z4, z8, 0);
}
#endif

// fallback: cooperative 64x64-half tile copy with same padded layout
__device__ __forceinline__ void stage_tile_coop(
    _Float16* dst, const _Float16* src, int src_stride, int tid)
{
    #pragma unroll
    for (int i = 0; i < 8; ++i) {
        int e = tid + i * 256;            // 2048 dwords total
        int row = e >> 5, c = e & 31;
        *(uint32_t*)(dst + row * TROW + c * 2) =
            *(const uint32_t*)(src + (size_t)row * src_stride + c * 2);
    }
}

// ---------------------------------------------------------------------------
// Kernel 1: projection GEMM  Y[8192,512] = X[8192,512] @ W[512,512]
// output written f16 in head-major layout; transpose_out=1 -> [b,h,d,s]
// ---------------------------------------------------------------------------
__global__ __launch_bounds__(256) void proj_kernel(
    const float* __restrict__ X, const float* __restrict__ W,
    _Float16* __restrict__ Out, int transpose_out)
{
    __shared__ _Float16 As[128][36];
    __shared__ _Float16 Bs[64][36];

    const int tid  = threadIdx.x;
    const int w    = tid >> 5, lane = tid & 31;
    const int hi   = lane >> 4, ln = lane & 15;
    const int wm   = w >> 1,   wn = w & 1;
    const int Mblk = blockIdx.x * 128;
    const int Nblk = blockIdx.y * 64;

    v8f acc[2][2] = {};

    for (int kt = 0; kt < DM_; kt += 32) {
        #pragma unroll
        for (int i = 0; i < 4; ++i) {
            int e = tid + i * 256;
            int row = e >> 3, c4 = (e & 7) * 4;
            float4 f = *(const float4*)(X + (size_t)(Mblk + row) * DM_ + kt + c4);
            As[row][c4 + 0] = (_Float16)f.x;  As[row][c4 + 1] = (_Float16)f.y;
            As[row][c4 + 2] = (_Float16)f.z;  As[row][c4 + 3] = (_Float16)f.w;
        }
        #pragma unroll
        for (int i = 0; i < 2; ++i) {
            int e = tid + i * 256;
            int kr = e >> 4, c4 = (e & 15) * 4;
            float4 f = *(const float4*)(W + (size_t)(kt + kr) * DM_ + Nblk + c4);
            Bs[c4 + 0][kr] = (_Float16)f.x;   Bs[c4 + 1][kr] = (_Float16)f.y;
            Bs[c4 + 2][kr] = (_Float16)f.z;   Bs[c4 + 3][kr] = (_Float16)f.w;
        }
        __syncthreads();

        Frag16 a[2], b[2];
        #pragma unroll
        for (int i = 0; i < 2; ++i) {
            int row = wm * 32 + i * 16 + ln;
            #pragma unroll
            for (int v = 0; v < 8; ++v)
                a[i].u[v] = *(const uint32_t*)&As[row][a_kidx(v, hi)];
        }
        #pragma unroll
        for (int j = 0; j < 2; ++j) {
            int col = wn * 32 + j * 16 + ln;
            #pragma unroll
            for (int v = 0; v < 8; ++v)
                b[j].u[v] = *(const uint32_t*)&Bs[col][2 * v + 16 * hi];
        }
        #pragma unroll
        for (int i = 0; i < 2; ++i)
            #pragma unroll
            for (int j = 0; j < 2; ++j)
                acc[i][j] = wmma_f16(a[i], b[j], acc[i][j]);
        __syncthreads();
    }

    #pragma unroll
    for (int i = 0; i < 2; ++i)
        #pragma unroll
        for (int j = 0; j < 2; ++j)
            #pragma unroll
            for (int r = 0; r < 8; ++r) {
                int m = Mblk + wm * 32 + i * 16 + r + 8 * hi;
                int n = Nblk + wn * 32 + j * 16 + ln;
                int bb = m >> 11, ss = m & (S_ - 1);
                int hh = n >> 6,  dd = n & 63;
                size_t idx = transpose_out
                    ? ((size_t)(bb * H_ + hh) * DKV_ + dd) * S_ + ss
                    : ((size_t)(bb * H_ + hh) * S_ + ss) * DKV_ + dd;
                Out[idx] = (_Float16)acc[i][j][r];
            }
}

// ---------------------------------------------------------------------------
// Kernel 2: attention. 1 workgroup = 128 q rows of one (b,h); 1 wave = 16 rows.
// K/V 64-key tiles staged in LDS by the Tensor Data Mover (double-buffered,
// wave0 issues DMA for tile t+1, waits TENSORcnt for tile t, barrier).
// Pass 1: WMMA QK^T -> online row max/sum. Pass 2: recompute QK^T, normalize,
// write attn once, relayout P through LDS, WMMA P@V from LDS V tile.
// ---------------------------------------------------------------------------
extern __shared__ char smem[];      // [K0|V0|K1|V1] 4*8448 B + P 9216 B

__global__ __launch_bounds__(256) void attn_kernel(
    const _Float16* __restrict__ Qh,   // [b,h,s,d]
    const _Float16* __restrict__ Kh,   // [b,h,s,d]
    const _Float16* __restrict__ Vt,   // [b,h,d,s]
    const unsigned char* __restrict__ mask, // [b,s,s] bool
    float* __restrict__ attn_out,      // [b,h,s,s]
    _Float16* __restrict__ Ctx)        // [b,s,h*dv]
{
    _Float16* Pl = (_Float16*)(smem + 4 * TILE_B);  // 8 waves x [16][36] halves

    const int tid = threadIdx.x, w = tid >> 5, lane = tid & 31;
    const int hi = lane >> 4, ln = lane & 15;
    const int bb = blockIdx.z, hh = blockIdx.y;
    const int bh = bb * H_ + hh;
    const int q0 = blockIdx.x * 128 + w * 16;

    const _Float16* Qb = Qh + (size_t)bh * S_ * DKV_;
    const _Float16* Kb = Kh + (size_t)bh * S_ * DKV_;
    const _Float16* Vb = Vt + (size_t)bh * DKV_ * S_;
    const unsigned char* mb = mask + (size_t)bb * S_ * S_;
    float* ab = attn_out + (size_t)bh * S_ * S_;

    // Q A-fragments (d = 0..31, 32..63), resident all kernel
    Frag16 aq[2];
    {
        const int row = q0 + ln;
        #pragma unroll
        for (int c = 0; c < 2; ++c)
            #pragma unroll
            for (int v = 0; v < 8; ++v)
                aq[c].u[v] = *(const uint32_t*)(Qb + (size_t)row * DKV_ + c * 32 + a_kidx(v, hi));
    }

    float rowmax[8], rowsum[8];
    #pragma unroll
    for (int r = 0; r < 8; ++r) { rowmax[r] = -3.0e38f; rowsum[r] = 0.0f; }

    const int NT = S_ / 64;

    // ---------------- pass 1: online softmax statistics ----------------
#if HAVE_TDM
    if (w == 0)
        tdm_load_2d(0u, Kb, 64, 64, 64, S_, 64);
#endif
    for (int t = 0; t < NT; ++t) {
        const int kt = t * 64;
#if HAVE_TDM
        if (w == 0) {
            if (t + 1 < NT) {
                tdm_load_2d((unsigned)((t + 1) & 1) * 2u * TILE_B,
                            Kb + (size_t)(kt + 64) * DKV_, 64, 64, 64, S_, 64);
                __builtin_amdgcn_s_wait_tensorcnt(1);
            } else {
                __builtin_amdgcn_s_wait_tensorcnt(0);
            }
        }
        __syncthreads();
        const _Float16* K0 = (const _Float16*)(smem + (t & 1) * (2 * TILE_B));
#else
        __syncthreads();
        stage_tile_coop((_Float16*)smem, Kb + (size_t)kt * DKV_, DKV_, tid);
        __syncthreads();
        const _Float16* K0 = (const _Float16*)smem;
#endif
        if (kt + 64 < S_)
            __builtin_prefetch(mb + (size_t)(q0 + ln) * S_ + kt + 64, 0, 1);

        #pragma unroll
        for (int st = 0; st < 4; ++st) {             // 16-key subtiles
            Frag16 bk[2];
            #pragma unroll
            for (int dc = 0; dc < 2; ++dc)
                #pragma unroll
                for (int v = 0; v < 8; ++v)
                    bk[dc].u[v] = *(const uint32_t*)&K0[(st * 16 + ln) * TROW + dc * 32 + 2 * v + 16 * hi];
            v8f sc = {};
            sc = wmma_f16(aq[0], bk[0], sc);
            sc = wmma_f16(aq[1], bk[1], sc);
            const int kk = kt + st * 16;
            #pragma unroll
            for (int r = 0; r < 8; ++r) {
                int m = r + 8 * hi;
                float s = sc[r] * SCALE;
                if (mb[(size_t)(q0 + m) * S_ + kk + ln]) s = -1e9f;
                float tm = s;
                #pragma unroll
                for (int o = 1; o < 16; o <<= 1) tm = fmaxf(tm, __shfl_xor(tm, o, 32));
                float mnew = fmaxf(rowmax[r], tm);
                float e = __expf(s - mnew);
                #pragma unroll
                for (int o = 1; o < 16; o <<= 1) e += __shfl_xor(e, o, 32);
                rowsum[r] = rowsum[r] * __expf(rowmax[r] - mnew) + e;
                rowmax[r] = mnew;
            }
        }
        __syncthreads();
    }
    float rinv[8];
    #pragma unroll
    for (int r = 0; r < 8; ++r) rinv[r] = rowsum[r] > 0.0f ? 1.0f / rowsum[r] : 0.0f;

    // ---------------- pass 2: normalize, emit attn, accumulate P@V ------
    v8f accv[4] = {};
#if HAVE_TDM
    if (w == 0) {
        tdm_load_2d(0u,     Kb, 64, 64, 64, S_, 64);
        tdm_load_2d(TILE_B, Vb, 64, 64, S_, 64, S_);
    }
#endif
    for (int t = 0; t < NT; ++t) {
        const int kt = t * 64;
#if HAVE_TDM
        if (w == 0) {
            if (t + 1 < NT) {
                unsigned boff = (unsigned)((t + 1) & 1) * 2u * TILE_B;
                tdm_load_2d(boff,          Kb + (size_t)(kt + 64) * DKV_, 64, 64, 64, S_, 64);
                tdm_load_2d(boff + TILE_B, Vb + (kt + 64),                64, 64, S_, 64, S_);
                __builtin_amdgcn_s_wait_tensorcnt(2);
            } else {
                __builtin_amdgcn_s_wait_tensorcnt(0);
            }
        }
        __syncthreads();
        const _Float16* K0 = (const _Float16*)(smem + (t & 1) * (2 * TILE_B));
        const _Float16* V0 = (const _Float16*)(smem + (t & 1) * (2 * TILE_B) + TILE_B);
#else
        __syncthreads();
        stage_tile_coop((_Float16*)smem, Kb + (size_t)kt * DKV_, DKV_, tid);
        stage_tile_coop((_Float16*)(smem + TILE_B), Vb + kt, S_, tid);
        __syncthreads();
        const _Float16* K0 = (const _Float16*)smem;
        const _Float16* V0 = (const _Float16*)(smem + TILE_B);
#endif
        #pragma unroll
        for (int c = 0; c < 2; ++c) {                // 32-key chunks
            #pragma unroll
            for (int hf = 0; hf < 2; ++hf) {         // 16-key subtiles
                const int stk = c * 32 + hf * 16;
                Frag16 bk[2];
                #pragma unroll
                for (int dc = 0; dc < 2; ++dc)
                    #pragma unroll
                    for (int v = 0; v < 8; ++v)
                        bk[dc].u[v] = *(const uint32_t*)&K0[(stk + ln) * TROW + dc * 32 + 2 * v + 16 * hi];
                v8f sc = {};
                sc = wmma_f16(aq[0], bk[0], sc);
                sc = wmma_f16(aq[1], bk[1], sc);
                #pragma unroll
                for (int r = 0; r < 8; ++r) {
                    int m = r + 8 * hi;
                    float s = sc[r] * SCALE;
                    if (mb[(size_t)(q0 + m) * S_ + kt + stk + ln]) s = -1e9f;
                    float p = __expf(s - rowmax[r]) * rinv[r];
                    ab[(size_t)(q0 + m) * S_ + kt + stk + ln] = p;     // attn, once
                    Pl[(w * 16 + m) * 36 + hf * 16 + ln] = (_Float16)p;
                }
            }
            // rebuild P as A-fragment (wave-private LDS, in-order DS ops)
            Frag16 pa;
            #pragma unroll
            for (int v = 0; v < 8; ++v)
                pa.u[v] = *(const uint32_t*)&Pl[(w * 16 + ln) * 36 + a_kidx(v, hi)];
            #pragma unroll
            for (int j = 0; j < 4; ++j) {
                Frag16 bv;
                #pragma unroll
                for (int v = 0; v < 8; ++v)
                    bv.u[v] = *(const uint32_t*)&V0[(j * 16 + ln) * TROW + c * 32 + 2 * v + 16 * hi];
                accv[j] = wmma_f16(pa, bv, accv[j]);
            }
        }
        __syncthreads();
    }

    // context -> [b, s, h*64 + d] f16
    #pragma unroll
    for (int j = 0; j < 4; ++j)
        #pragma unroll
        for (int r = 0; r < 8; ++r) {
            int m = r + 8 * hi;
            int ss = q0 + m;
            Ctx[((size_t)bb * S_ + ss) * DM_ + hh * 64 + j * 16 + ln] = (_Float16)accv[j][r];
        }
}

// ---------------------------------------------------------------------------
// Kernel 3: output projection + residual. Tmp = Ctx(f16) @ W_O + input_Q
// ---------------------------------------------------------------------------
__global__ __launch_bounds__(256) void outproj_kernel(
    const _Float16* __restrict__ Ctx, const float* __restrict__ W,
    const float* __restrict__ resid, float* __restrict__ Tmp)
{
    __shared__ _Float16 As[128][36];
    __shared__ _Float16 Bs[64][36];

    const int tid  = threadIdx.x;
    const int w    = tid >> 5, lane = tid & 31;
    const int hi   = lane >> 4, ln = lane & 15;
    const int wm   = w >> 1,   wn = w & 1;
    const int Mblk = blockIdx.x * 128;
    const int Nblk = blockIdx.y * 64;

    v8f acc[2][2] = {};

    for (int kt = 0; kt < DM_; kt += 32) {
        #pragma unroll
        for (int i = 0; i < 4; ++i) {
            int e = tid + i * 256;
            int row = e >> 3, c = (e & 7) * 4;
            uint2 q = *(const uint2*)(Ctx + (size_t)(Mblk + row) * DM_ + kt + c);
            *(uint2*)&As[row][c] = q;
        }
        #pragma unroll
        for (int i = 0; i < 2; ++i) {
            int e = tid + i * 256;
            int kr = e >> 4, c4 = (e & 15) * 4;
            float4 f = *(const float4*)(W + (size_t)(kt + kr) * DM_ + Nblk + c4);
            Bs[c4 + 0][kr] = (_Float16)f.x;  Bs[c4 + 1][kr] = (_Float16)f.y;
            Bs[c4 + 2][kr] = (_Float16)f.z;  Bs[c4 + 3][kr] = (_Float16)f.w;
        }
        __syncthreads();

        Frag16 a[2], b[2];
        #pragma unroll
        for (int i = 0; i < 2; ++i) {
            int row = wm * 32 + i * 16 + ln;
            #pragma unroll
            for (int v = 0; v < 8; ++v)
                a[i].u[v] = *(const uint32_t*)&As[row][a_kidx(v, hi)];
        }
        #pragma unroll
        for (int j = 0; j < 2; ++j) {
            int col = wn * 32 + j * 16 + ln;
            #pragma unroll
            for (int v = 0; v < 8; ++v)
                b[j].u[v] = *(const uint32_t*)&Bs[col][2 * v + 16 * hi];
        }
        #pragma unroll
        for (int i = 0; i < 2; ++i)
            #pragma unroll
            for (int j = 0; j < 2; ++j)
                acc[i][j] = wmma_f16(a[i], b[j], acc[i][j]);
        __syncthreads();
    }

    #pragma unroll
    for (int i = 0; i < 2; ++i)
        #pragma unroll
        for (int j = 0; j < 2; ++j)
            #pragma unroll
            for (int r = 0; r < 8; ++r) {
                int m = Mblk + wm * 32 + i * 16 + r + 8 * hi;
                int n = Nblk + wn * 32 + j * 16 + ln;
                size_t idx = (size_t)m * DM_ + n;
                Tmp[idx] = acc[i][j][r] + resid[idx];
            }
}

// ---------------------------------------------------------------------------
// Kernel 4: row LayerNorm over 512 columns. One block per row.
// ---------------------------------------------------------------------------
__global__ __launch_bounds__(256) void ln_kernel(
    const float* __restrict__ Tmp, float* __restrict__ Out)
{
    __shared__ float red0[8];
    __shared__ float red1[8];
    const int row = blockIdx.x, tid = threadIdx.x;
    const size_t base = (size_t)row * DM_;

    float x0 = Tmp[base + tid];
    float x1 = Tmp[base + 256 + tid];

    float s = x0 + x1;
    #pragma unroll
    for (int o = 1; o < 32; o <<= 1) s += __shfl_xor(s, o, 32);
    if ((tid & 31) == 0) red0[tid >> 5] = s;
    __syncthreads();
    float mean = 0.0f;
    #pragma unroll
    for (int i = 0; i < 8; ++i) mean += red0[i];
    mean *= (1.0f / (float)DM_);

    float d0 = x0 - mean, d1 = x1 - mean;
    float vs = d0 * d0 + d1 * d1;
    #pragma unroll
    for (int o = 1; o < 32; o <<= 1) vs += __shfl_xor(vs, o, 32);
    if ((tid & 31) == 0) red1[tid >> 5] = vs;
    __syncthreads();
    float var = 0.0f;
    #pragma unroll
    for (int i = 0; i < 8; ++i) var += red1[i];
    var *= (1.0f / (float)DM_);

    float rs = rsqrtf(var + LN_EPS);
    Out[base + tid]       = d0 * rs;
    Out[base + 256 + tid] = d1 * rs;
}

// ---------------------------------------------------------------------------
extern "C" void kernel_launch(void* const* d_in, const int* in_sizes, int n_in,
                              void* d_out, int out_size, void* d_ws, size_t ws_size,
                              hipStream_t stream) {
    (void)in_sizes; (void)n_in; (void)out_size; (void)ws_size;

    const float* inQ = (const float*)d_in[0];
    const float* inK = (const float*)d_in[1];
    const float* inV = (const float*)d_in[2];
    const unsigned char* mask = (const unsigned char*)d_in[3];   // jnp.bool_
    const float* wq  = (const float*)d_in[4];
    const float* wk  = (const float*)d_in[5];
    const float* wv  = (const float*)d_in[6];
    const float* wo  = (const float*)d_in[7];

    char* ws = (char*)d_ws;                                   // 48 MB used
    _Float16* Qh  = (_Float16*)(ws + ((size_t)0  << 20));     //  8 MB [b,h,s,d]
    _Float16* Kh  = (_Float16*)(ws + ((size_t)8  << 20));     //  8 MB [b,h,s,d]
    _Float16* Vt  = (_Float16*)(ws + ((size_t)16 << 20));     //  8 MB [b,h,d,s]
    _Float16* Ctx = (_Float16*)(ws + ((size_t)24 << 20));     //  8 MB [b,s,512]
    float*    Tmp = (float*)   (ws + ((size_t)32 << 20));     // 16 MB [b,s,512]

    float* out  = (float*)d_out;
    float* attn = out + (size_t)B_ * S_ * DM_;

    dim3 gproj(8192 / 128, DM_ / 64);
    proj_kernel<<<gproj, 256, 0, stream>>>(inQ, wq, Qh, 0);
    proj_kernel<<<gproj, 256, 0, stream>>>(inK, wk, Kh, 0);
    proj_kernel<<<gproj, 256, 0, stream>>>(inV, wv, Vt, 1);

    // dynamic LDS: K0|V0|K1|V1 (4 * 8448 B) + P relayout (8*16*36*2 B)
    const size_t attn_lds = 4 * (size_t)TILE_B + 8 * 16 * 36 * sizeof(_Float16);
    dim3 gattn(S_ / 128, H_, B_);
    attn_kernel<<<gattn, 256, attn_lds, stream>>>(Qh, Kh, Vt, mask, attn, Ctx);

    outproj_kernel<<<gproj, 256, 0, stream>>>(Ctx, wo, inQ, Tmp);
    ln_kernel<<<B_ * S_, 256, 0, stream>>>(Tmp, out);
}
